// SpiralAutoencoder_24627342475494
// MI455X (gfx1250) — matrix operand order repaired and checked by reading, not compiled, not run on hardware
//
#include <hip/hip_runtime.h>
#include <hip/hip_bf16.h>

// ---------------------------------------------------------------------------
// MI455X (gfx1250) spiral autoencoder forward pass.
// All GEMM-shaped work runs on v_wmma_f32_16x16x32_f16 (f16 operands staged
// into LDS from f32, f32 accumulate). Base tile: 128x32 block, K-stage 64;
// 8 waves in a 4x2 grid, each wave owns a 32x16 output (2 accumulators).
// The batch-shared resample GEMM (dominant op: U[0] 5024x1257) additionally
// tiles NB=4 batches per block so each staged A-tile feeds 16 WMMAs.
// ---------------------------------------------------------------------------

typedef __attribute__((ext_vector_type(16))) _Float16 v16h;
typedef __attribute__((ext_vector_type(8)))  float    v8f;

// Fragment loaders (wave32), contiguous k = 16h + s in both A and B (not the ISA table order):
//   A 16x32: lanes 0-15 row M=lane, VGPR0-7 K=0..15;
//            lanes 16-31 same rows, K=16..31.
//   B 32x16: lanes 0-15 col N=lane, VGPR v -> K=2v,2v+1; lanes 16-31 K=16+.
__device__ __forceinline__ v16h load_a_frag(const _Float16* A, int lda)
{
  const int l  = threadIdx.x & 31;
  const int ar = l & 15;
  const int ak = (l & 16) ? 16 : 0;
  v16h av;
#pragma unroll
  for (int v = 0; v < 4; ++v) {
    av[2 * v]         = A[ar * lda + ak + 2 * v];
    av[2 * v + 1]     = A[ar * lda + ak + 2 * v + 1];
    av[8 + 2 * v]     = A[ar * lda + 8 + ak + 2 * v];
    av[8 + 2 * v + 1] = A[ar * lda + 8 + ak + 2 * v + 1];
  }
  return av;
}

__device__ __forceinline__ v16h load_b_frag(const _Float16* B, int ldb)
{
  const int l  = threadIdx.x & 31;
  const int bc = l & 15;
  const int bk = (l & 16) ? 16 : 0;
  v16h bv;
#pragma unroll
  for (int v = 0; v < 8; ++v) {
    bv[2 * v]     = B[(bk + 2 * v) * ldb + bc];
    bv[2 * v + 1] = B[(bk + 2 * v + 1) * ldb + bc];
  }
  return bv;
}

#define WMMA_F16(a, b, c) \
  __builtin_amdgcn_wmma_f32_16x16x32_f16(false, (a), false, (b), (short)0, (c), false, false)

// ---------------------------------------------------------------------------
// Spiral conv as WMMA GEMM, gather fused into the A-tile load:
//   out(b,n,o) = act( sum_{s,c} x(b, sp(n,s), c) * W(o, s*Cin+c) + bias(o) )
// then out(:, Nv-1, :) = 0. Each thread owns ONE A-row (row decode hoisted,
// single integer divide) and 32 contiguous K columns; the (s,c) split is
// maintained incrementally -> no divides in the K loop.
// ---------------------------------------------------------------------------
__global__ __launch_bounds__(256)
void k_sconv_wmma(const float* __restrict__ x, const int* __restrict__ sp,
                  const float* __restrict__ Wm, const float* __restrict__ bias,
                  float* __restrict__ out,
                  int Bn, int Nv, int Cin, int Cout, int act)
{
  const int Kdim = 9 * Cin;
  const int Mtot = Bn * Nv;
  __shared__ _Float16 As[128 * 64];   // 16 KB
  __shared__ _Float16 Bs[64 * 32];    //  4 KB
  const int tid  = threadIdx.x;
  const int m0   = blockIdx.x * 128;
  const int n0   = blockIdx.y * 32;
  const int wave = tid >> 5;
  const int wm   = wave >> 1;   // 0..3 -> rows wm*32..wm*32+31
  const int wn   = wave & 1;    // 0..1 -> cols wn*16..wn*16+15

  // This thread's fixed A-row (2 threads per row, 32 cols each).
  const int arow    = tid >> 1;          // 0..127
  const int colbase = (tid & 1) * 32;    // 0 or 32
  const int gm      = m0 + arow;
  const bool mrow   = (gm < Mtot);
  int bb = 0, nn = 0;
  if (mrow) { bb = gm / Nv; nn = gm - bb * Nv; }
  const int*   sprow = sp + nn * 9;
  const float* xb    = x + (size_t)bb * Nv * Cin;

  v8f acc0 = {}, acc1 = {};
  for (int k0 = 0; k0 < Kdim; k0 += 64) {
    // ---- stage A (gathered, f32->f16), incremental (s,c) split ----
    {
      int gk = k0 + colbase;
      int s  = gk / Cin;
      int c  = gk - s * Cin;
#pragma unroll
      for (int i = 0; i < 32; ++i) {
        float val = 0.f;
        if (mrow && (gk + i) < Kdim) val = xb[sprow[s] * Cin + c];
        As[arow * 64 + colbase + i] = (_Float16)val;
        if (++c == Cin) { c = 0; ++s; }
      }
    }
    // ---- stage B: W(Cout, Kdim) transposed tile ----
#pragma unroll
    for (int i = 0; i < 8; ++i) {
      int idx = tid + i * 256;
      int kk = idx >> 5, cnn = idx & 31;
      int gk = k0 + kk, gn = n0 + cnn;
      Bs[kk * 32 + cnn] =
          (_Float16)((gk < Kdim && gn < Cout) ? Wm[gn * Kdim + gk] : 0.f);
    }
    if (k0 + 64 < Kdim)
      __builtin_prefetch(&Wm[(n0 + (tid & 31)) * Kdim + k0 + 64], 0, 1);
    __syncthreads();
    // ---- 4 WMMAs per wave per stage; B fragment reused across M-tiles ----
#pragma unroll
    for (int ks = 0; ks < 64; ks += 32) {
      v16h bf = load_b_frag(&Bs[ks * 32 + wn * 16], 32);
      v16h a0 = load_a_frag(&As[(wm * 32) * 64 + ks], 64);
      v16h a1 = load_a_frag(&As[(wm * 32 + 16) * 64 + ks], 64);
      acc0 = WMMA_F16(a0, bf, acc0);
      acc1 = WMMA_F16(a1, bf, acc1);
    }
    __syncthreads();
  }
  // ---- epilogue: bias + ELU + pad-vertex mask ----
  const int l = tid & 31;
  const int col = l & 15, r0 = (l & 16) ? 8 : 0;
  const int gn = n0 + wn * 16 + col;
#pragma unroll
  for (int v = 0; v < 8; ++v) {
    int g0 = m0 + wm * 32 + r0 + v;
    int g1 = g0 + 16;
    if (gn < Cout) {
      if (g0 < Mtot) {
        float val = acc0[v] + bias[gn];
        if (act) val = (val > 0.f) ? val : (__expf(val) - 1.f);
        if ((g0 % Nv) == Nv - 1) val = 0.f;
        out[g0 * Cout + gn] = val;
      }
      if (g1 < Mtot) {
        float val = acc1[v] + bias[gn];
        if (act) val = (val > 0.f) ? val : (__expf(val) - 1.f);
        if ((g1 % Nv) == Nv - 1) val = 0.f;
        out[g1 * Cout + gn] = val;
      }
    }
  }
}

// ---------------------------------------------------------------------------
// Batch-tiled dense resample: out[b] (Mo x C) = Mmat (Mo x Ni) @ X[b] (Ni x C)
// for NB=4 batches per block. The batch-invariant Mmat tile is staged ONCE
// per K-stage and its fragments reused across 4 batch B-tiles -> 16 WMMAs
// per wave per stage, 4x less L2 traffic on the dominant U[0]/D[0] matrices.
// ---------------------------------------------------------------------------
#define NB 4
__global__ __launch_bounds__(256)
void k_resample_wmma(const float* __restrict__ Mmat, const float* __restrict__ X,
                     float* __restrict__ out, int Mo, int Ni, int C)
{
  const int b0 = blockIdx.z * NB;
  __shared__ _Float16 As[128 * 64];        // 16 KB (shared across batches)
  __shared__ _Float16 Bs[NB][64 * 32];     // 16 KB (one tile per batch)
  const int tid  = threadIdx.x;
  const int m0   = blockIdx.x * 128;
  const int n0   = blockIdx.y * 32;
  const int wave = tid >> 5;
  const int wm   = wave >> 1;
  const int wn   = wave & 1;
  const int arow    = tid >> 1;
  const int colbase = (tid & 1) * 32;
  const int gm      = m0 + arow;
  const bool mrow   = (gm < Mo);
  const float* arp  = Mmat + (size_t)gm * Ni;

  v8f acc0[NB], acc1[NB];
#pragma unroll
  for (int nb = 0; nb < NB; ++nb) { acc0[nb] = (v8f){}; acc1[nb] = (v8f){}; }

  for (int k0 = 0; k0 < Ni; k0 += 64) {
    // ---- stage shared A tile (f32->f16) ----
#pragma unroll
    for (int i = 0; i < 32; ++i) {
      int gk = k0 + colbase + i;
      As[arow * 64 + colbase + i] =
          (_Float16)((mrow && gk < Ni) ? arp[gk] : 0.f);
    }
    // ---- stage NB batch B tiles ----
#pragma unroll
    for (int nb = 0; nb < NB; ++nb) {
      const float* Xb = X + (size_t)(b0 + nb) * Ni * C;
#pragma unroll
      for (int i = 0; i < 8; ++i) {
        int idx = tid + i * 256;
        int kk = idx >> 5, cnn = idx & 31;
        int gk = k0 + kk, gn = n0 + cnn;
        Bs[nb][kk * 32 + cnn] =
            (_Float16)((gk < Ni && gn < C) ? Xb[gk * C + gn] : 0.f);
      }
    }
    if (k0 + 64 < Ni)
      __builtin_prefetch(&arp[k0 + 64 + colbase], 0, 1);
    __syncthreads();
    // ---- 16 WMMAs per wave per stage; A fragments reused across batches ----
#pragma unroll
    for (int ks = 0; ks < 64; ks += 32) {
      v16h a0 = load_a_frag(&As[(wm * 32) * 64 + ks], 64);
      v16h a1 = load_a_frag(&As[(wm * 32 + 16) * 64 + ks], 64);
#pragma unroll
      for (int nb = 0; nb < NB; ++nb) {
        v16h bf = load_b_frag(&Bs[nb][ks * 32 + wn * 16], 32);
        acc0[nb] = WMMA_F16(a0, bf, acc0[nb]);
        acc1[nb] = WMMA_F16(a1, bf, acc1[nb]);
      }
    }
    __syncthreads();
  }
  const int l = tid & 31;
  const int col = l & 15, r0 = (l & 16) ? 8 : 0;
  const int gn = n0 + wn * 16 + col;
#pragma unroll
  for (int nb = 0; nb < NB; ++nb) {
    float* outb = out + (size_t)(b0 + nb) * Mo * C;
#pragma unroll
    for (int v = 0; v < 8; ++v) {
      int g0 = m0 + wm * 32 + r0 + v;
      int g1 = g0 + 16;
      if (gn < C) {
        if (g0 < Mo) outb[g0 * C + gn] = acc0[nb][v];
        if (g1 < Mo) outb[g1 * C + gn] = acc1[nb][v];
      }
    }
  }
}

// ---------------------------------------------------------------------------
// Generic NT GEMM: out(M,N) = act( A(M,K) @ W(N,K)^T + bias )  (FC, attn proj)
// ---------------------------------------------------------------------------
__global__ __launch_bounds__(256)
void k_gemm_nt_wmma(const float* __restrict__ A, const float* __restrict__ Wm,
                    const float* __restrict__ bias, float* __restrict__ out,
                    int M, int N, int K, int act)
{
  __shared__ _Float16 As[128 * 64];
  __shared__ _Float16 Bs[64 * 32];
  const int tid  = threadIdx.x;
  const int m0   = blockIdx.x * 128;
  const int n0   = blockIdx.y * 32;
  const int wave = tid >> 5;
  const int wm   = wave >> 1;
  const int wn   = wave & 1;
  const int arow    = tid >> 1;
  const int colbase = (tid & 1) * 32;
  const int gm      = m0 + arow;
  const bool mrow   = (gm < M);
  const float* arp  = A + (size_t)gm * K;

  v8f acc0 = {}, acc1 = {};
  for (int k0 = 0; k0 < K; k0 += 64) {
#pragma unroll
    for (int i = 0; i < 32; ++i) {
      int gk = k0 + colbase + i;
      As[arow * 64 + colbase + i] =
          (_Float16)((mrow && gk < K) ? arp[gk] : 0.f);
    }
#pragma unroll
    for (int i = 0; i < 8; ++i) {
      int idx = tid + i * 256;
      int kk = idx >> 5, cnn = idx & 31;
      int gk = k0 + kk, gn = n0 + cnn;
      Bs[kk * 32 + cnn] =
          (_Float16)((gk < K && gn < N) ? Wm[gn * K + gk] : 0.f);
    }
    __syncthreads();
#pragma unroll
    for (int ks = 0; ks < 64; ks += 32) {
      v16h bf = load_b_frag(&Bs[ks * 32 + wn * 16], 32);
      v16h a0 = load_a_frag(&As[(wm * 32) * 64 + ks], 64);
      v16h a1 = load_a_frag(&As[(wm * 32 + 16) * 64 + ks], 64);
      acc0 = WMMA_F16(a0, bf, acc0);
      acc1 = WMMA_F16(a1, bf, acc1);
    }
    __syncthreads();
  }
  const int l = tid & 31;
  const int col = l & 15, r0 = (l & 16) ? 8 : 0;
  const int gn = n0 + wn * 16 + col;
#pragma unroll
  for (int v = 0; v < 8; ++v) {
    int g0 = m0 + wm * 32 + r0 + v;
    int g1 = g0 + 16;
    if (gn < N) {
      if (g0 < M) {
        float val = acc0[v] + bias[gn];
        if (act) val = (val > 0.f) ? val : (__expf(val) - 1.f);
        out[g0 * N + gn] = val;
      }
      if (g1 < M) {
        float val = acc1[v] + bias[gn];
        if (act) val = (val > 0.f) ? val : (__expf(val) - 1.f);
        out[g1 * N + gn] = val;
      }
    }
  }
}

// ---------------------------------------------------------------------------
// Attention core: softmax(q k^T * hd^-0.5) v per (batch, head, query row).
// hd <= 16. Max-subtracted softmax (matches jax.nn.softmax).
// ---------------------------------------------------------------------------
__global__ void k_attn_core(const float* __restrict__ q,
                            const float* __restrict__ k,
                            const float* __restrict__ v,
                            float* __restrict__ out, int Bn, int L, int E, int H)
{
  int idx = blockIdx.x * blockDim.x + threadIdx.x;
  int total = Bn * H * L;
  if (idx >= total) return;
  int qi = idx % L;
  int t = idx / L;
  int h = t % H;
  int b = t / H;
  int hd = E / H;
  float scale = rsqrtf((float)hd);
  const float* qp = q + ((size_t)b * L + qi) * E + h * hd;
  float maxs = -1e30f;
  for (int j = 0; j < L; ++j) {
    const float* kp = k + ((size_t)b * L + j) * E + h * hd;
    float s = 0.f;
    for (int d = 0; d < hd; ++d) s += qp[d] * kp[d];
    maxs = fmaxf(maxs, s * scale);
  }
  float accv[16];
  for (int d = 0; d < hd; ++d) accv[d] = 0.f;
  float denom = 0.f;
  for (int j = 0; j < L; ++j) {
    const float* kp = k + ((size_t)b * L + j) * E + h * hd;
    float s = 0.f;
    for (int d = 0; d < hd; ++d) s += qp[d] * kp[d];
    float w = __expf(s * scale - maxs);
    denom += w;
    const float* vp = v + ((size_t)b * L + j) * E + h * hd;
    for (int d = 0; d < hd; ++d) accv[d] += w * vp[d];
  }
  float* op = out + ((size_t)b * L + qi) * E + h * hd;
  float inv = 1.f / denom;
  for (int d = 0; d < hd; ++d) op[d] = accv[d] * inv;
}

// out[i] = a[i] + s[sidx] * b[i]   (scale read on-device -> deterministic)
__global__ void k_axpy(const float* __restrict__ a, const float* __restrict__ b,
                       const float* __restrict__ s, int sidx,
                       float* __restrict__ o, int n)
{
  int i = blockIdx.x * blockDim.x + threadIdx.x;
  if (i < n) o[i] = a[i] + s[sidx] * b[i];
}

__global__ void k_finalize(const float* __restrict__ x2,
                           const float* __restrict__ neutral,
                           float* __restrict__ o1, float* __restrict__ o2, int n)
{
  int i = blockIdx.x * blockDim.x + threadIdx.x;
  if (i < n) {
    float v = x2[i];
    o1[i] = v;
    o2[i] = v + neutral[i];
  }
}

// Deterministic two-pass L1 reduction (no float atomics).
__global__ void k_loss_partial(const float* __restrict__ a,
                               const float* __restrict__ b,
                               float* __restrict__ partial, int n)
{
  __shared__ float red[256];
  float s = 0.f;
  for (int i = blockIdx.x * 256 + threadIdx.x; i < n; i += gridDim.x * 256)
    s += fabsf(a[i] - b[i]);
  red[threadIdx.x] = s;
  __syncthreads();
  for (int st = 128; st > 0; st >>= 1) {
    if (threadIdx.x < st) red[threadIdx.x] += red[threadIdx.x + st];
    __syncthreads();
  }
  if (threadIdx.x == 0) partial[blockIdx.x] = red[0];
}

__global__ void k_loss_final(const float* __restrict__ partial, int np,
                             const float* __restrict__ ploss,
                             float* __restrict__ out, float invn)
{
  __shared__ float red[256];
  float s = 0.f;
  for (int i = threadIdx.x; i < np; i += 256) s += partial[i];
  red[threadIdx.x] = s;
  __syncthreads();
  for (int st = 128; st > 0; st >>= 1) {
    if (threadIdx.x < st) red[threadIdx.x] += red[threadIdx.x + st];
    __syncthreads();
  }
  if (threadIdx.x == 0) out[0] = red[0] * invn * ploss[0];
}

// ---------------------------------------------------------------------------
// Host orchestration. Input flattening assumption: top-level dict order
// {landmarks_dis, landmark_n, neutral_points, params, spirals}; 'params'
// tree-flattened with dict keys sorted (jax convention): D[0..3], U[0..3],
// att[0](4), att[1](4), d1[0..4](W,b), d2[0..4](W,b), enc[0..3](W,b),
// fc_dec(W,b), fc_dec1(W,b), lambda_atten, para, para_atten, para_loss;
// then spirals[0..3].
// ---------------------------------------------------------------------------
extern "C" void kernel_launch(void* const* d_in, const int* in_sizes, int n_in,
                              void* d_out, int out_size, void* d_ws, size_t ws_size,
                              hipStream_t stream)
{
  (void)in_sizes; (void)n_in; (void)out_size; (void)ws_size;
  auto F = [&](int i) { return (const float*)d_in[i]; };
  auto I = [&](int i) { return (const int*)d_in[i]; };

  const float* landmarks  = F(0);
  const float* landmarkN  = F(1);
  const float* neutral    = F(2);
  const float* Dm[4]      = {F(3), F(4), F(5), F(6)};
  const float* Um[4]      = {F(7), F(8), F(9), F(10)};
  const float* attW[2][4] = {{F(11), F(12), F(13), F(14)},
                             {F(15), F(16), F(17), F(18)}};
  const float* d1W[5] = {F(19), F(21), F(23), F(25), F(27)};
  const float* d1B[5] = {F(20), F(22), F(24), F(26), F(28)};
  const float* d2W[5] = {F(29), F(31), F(33), F(35), F(37)};
  const float* d2B[5] = {F(30), F(32), F(34), F(36), F(38)};
  const float* encW[4] = {F(39), F(41), F(43), F(45)};
  const float* encB[4] = {F(40), F(42), F(44), F(46)};
  const float* fcW = F(47),  *fcB  = F(48);
  const float* fc1W = F(49), *fc1B = F(50);
  const float* lam      = F(51);
  const float* para     = F(52);
  const float* paraAtt  = F(53);
  const float* paraLoss = F(54);
  const int* Sp[4] = {I(55), I(56), I(57), I(58)};

  const int NP[5]  = {5024, 1257, 315, 80, 21};
  const int FDc[6] = {128, 64, 32, 32, 16, 3};
  const int Bn = 32;

  // ---- workspace carve-out (~100 MB of f32) ----
  float* ws = (float*)d_ws;
  size_t off = 0;
  auto alloc = [&](size_t n) {
    float* p = ws + off;
    off += (n + 63) & ~(size_t)63;
    return p;
  };
  float* etmp  = alloc((size_t)Bn * 5024 * 16);
  float* Xs[4] = {alloc((size_t)Bn * 1257 * 16), alloc((size_t)Bn * 315 * 32),
                  alloc((size_t)Bn * 80 * 64),  alloc((size_t)Bn * 21 * 128)};
  float* xd = alloc((size_t)Bn * 2688);
  float* xn = alloc((size_t)Bn * 2688);
  const size_t big = (size_t)Bn * 5024 * 32;
  float* bufA = alloc(big);
  float* bufB = alloc(big);
  float* bufC = alloc(big);
  float* bufD = alloc(big);
  const size_t attbig = (size_t)Bn * 315 * 32;
  float* qb = alloc(attbig);
  float* kb = alloc(attbig);
  float* vb = alloc(attbig);
  float* cb = alloc(attbig);
  float* pb = alloc(attbig);
  float* partial = alloc(512);

  auto sconv = [&](const float* x, const int* sp, const float* Wm,
                   const float* bias, float* o, int Nv, int Cin, int Cout, int act) {
    dim3 g((unsigned)((Bn * Nv + 127) / 128), (unsigned)((Cout + 31) / 32));
    k_sconv_wmma<<<g, 256, 0, stream>>>(x, sp, Wm, bias, o, Bn, Nv, Cin, Cout, act);
  };
  auto resample = [&](const float* Mm, const float* X, float* o, int Mo, int Ni, int C) {
    dim3 g((unsigned)((Mo + 127) / 128), (unsigned)((C + 31) / 32), (unsigned)(Bn / NB));
    k_resample_wmma<<<g, 256, 0, stream>>>(Mm, X, o, Mo, Ni, C);
  };
  auto gemm = [&](const float* A, const float* Wm, const float* bias, float* o,
                  int M, int N, int K, int act) {
    dim3 g((unsigned)((M + 127) / 128), (unsigned)((N + 31) / 32));
    k_gemm_nt_wmma<<<g, 256, 0, stream>>>(A, Wm, bias, o, M, N, K, act);
  };
  auto axpy = [&](const float* a, const float* b, const float* s, int sidx,
                  float* o, int n) {
    k_axpy<<<(n + 255) / 256, 256, 0, stream>>>(a, b, s, sidx, o, n);
  };

  // ---- encoder ----
  sconv(neutral, Sp[0], encW[0], encB[0], etmp, 5024, 3, 16, 1);
  resample(Dm[0], etmp, Xs[0], 1257, 5024, 16);
  sconv(Xs[0], Sp[1], encW[1], encB[1], etmp, 1257, 16, 32, 1);
  resample(Dm[1], etmp, Xs[1], 315, 1257, 32);
  sconv(Xs[1], Sp[2], encW[2], encB[2], etmp, 315, 32, 64, 1);
  resample(Dm[2], etmp, Xs[2], 80, 315, 64);
  sconv(Xs[2], Sp[3], encW[3], encB[3], etmp, 80, 64, 128, 1);
  resample(Dm[3], etmp, Xs[3], 21, 80, 128);

  // ---- decoder FC heads ----
  gemm(landmarks, fcW, fcB, xd, Bn, 2688, 204, 0);
  gemm(landmarkN, fc1W, fc1B, xn, Bn, 2688, 204, 0);
  axpy(xd, xn, lam, 0, bufA, Bn * 2688);   // x1 = xd + lambda*xn

  float* x1c = bufA;
  float* x1o = bufB;
  auto x1swap = [&] { float* t = x1c; x1c = x1o; x1o = t; };
  const float* x2c = xd;
  float* x2buf[2] = {bufC, bufD};
  int x2i = 0;
  auto x2out = [&] { float* r = x2buf[x2i]; x2i ^= 1; return r; };

  // ---- decoder levels ----
  for (int i = 0; i < 4; ++i) {
    const int lvl = 3 - i;
    const int Lin = NP[lvl + 1], Lout = NP[lvl];
    const int Cc = FDc[i], Co = FDc[i + 1];
    if (i < 3) {  // x2 skip connection
      float* o = x2out();
      axpy(x2c, Xs[lvl], para, i, o, Bn * Lin * Cc);
      x2c = o;
    }
    resample(Um[lvl], x1c, x1o, Lout, Lin, Cc); x1swap();
    { float* o = x2out(); resample(Um[lvl], x2c, o, Lout, Lin, Cc); x2c = o; }
    sconv(x1c, Sp[lvl], d1W[i], d1B[i], x1o, Lout, Cc, Co, 1); x1swap();
    { float* o = x2out(); sconv(x2c, Sp[lvl], d2W[i], d2B[i], o, Lout, Cc, Co, 1); x2c = o; }
    if (i < 2) {  // MHA(x1, x2) -> residual into x1
      const int E = Co, L = Lout, M = Bn * L;
      gemm(x1c, attW[i][0],             attW[i][1],         qb, M, E, E, 0);
      gemm(x2c, attW[i][0] + E * E,     attW[i][1] + E,     kb, M, E, E, 0);
      gemm(x2c, attW[i][0] + 2 * E * E, attW[i][1] + 2 * E, vb, M, E, E, 0);
      const int tot = Bn * 4 * L;
      k_attn_core<<<(tot + 255) / 256, 256, 0, stream>>>(qb, kb, vb, cb, Bn, L, E, 4);
      gemm(cb, attW[i][2], attW[i][3], pb, M, E, E, 0);
      axpy(x1c, pb, paraAtt, i + 1, x1o, M * E); x1swap();
    }
  }

  // ---- final convs (identity activation, Cout=3) ----
  sconv(x1c, Sp[0], d1W[4], d1B[4], x1o, 5024, 16, 3, 0); x1swap();
  { float* o = x2out(); sconv(x2c, Sp[0], d2W[4], d2B[4], o, 5024, 16, 3, 0); x2c = o; }

  // ---- outputs: x2, x2+neutral, loss ----
  float* out = (float*)d_out;
  const int n3 = Bn * 5024 * 3;
  k_finalize<<<(n3 + 255) / 256, 256, 0, stream>>>(x2c, neutral, out, out + n3, n3);
  k_loss_partial<<<512, 256, 0, stream>>>(x1c, x2c, partial, n3);
  k_loss_final<<<1, 256, 0, stream>>>(partial, 512, paraLoss, out + 2 * n3, 1.0f / n3);
}
